// MemMambaBlock_88965952569957
// MI455X (gfx1250) — compile-verified, matrix operand order checked by reading, not run on hardware
//
#include <hip/hip_runtime.h>
#include <hip/hip_bf16.h>

// ---------------------------------------------------------------------------
// MemMamba block for gfx1250 (MI455X). GEMMs via v_wmma_f32_16x16x32_bf16,
// scan / pool / attention via wave32 VALU kernels.
// ---------------------------------------------------------------------------

#define D_MODEL   1024
#define D_STATE   128
#define D_CONV    4
#define D_INNER   2048
#define NHEADS    32
#define HEADDIM   64
#define POOLN     64
#define SUMDIM    256
#define BB        4
#define TT        2048
#define BT        (BB*TT)
#define D_INPROJ  (2*D_INNER + 2*D_STATE + NHEADS)   // 4384
#define XBC_DIM   (D_INNER + 2*D_STATE)              // 2304

typedef __attribute__((ext_vector_type(16))) __bf16 v16bf;
typedef __attribute__((ext_vector_type(8)))  __bf16 v8bf;
typedef __attribute__((ext_vector_type(8)))  float  v8f;

__device__ __forceinline__ unsigned short f32_to_bf16u(float f) {
    unsigned int u = __float_as_uint(f);
    u += 0x7fffu + ((u >> 16) & 1u);       // round-to-nearest-even
    return (unsigned short)(u >> 16);
}
__device__ __forceinline__ float bf16u_to_f32(unsigned short h) {
    return __uint_as_float(((unsigned int)h) << 16);
}
__device__ __forceinline__ float sigmoidf_(float x) { return 1.f / (1.f + __expf(-x)); }
__device__ __forceinline__ float siluf_(float x)    { return x * sigmoidf_(x); }

// ---------------------------------------------------------------------------
// Generic convert f32 -> bf16 (contiguous)
// ---------------------------------------------------------------------------
__global__ void cvt_f32_bf16(const float* __restrict__ src, unsigned short* __restrict__ dst, long n) {
    long i = (long)blockIdx.x * blockDim.x + threadIdx.x;
    if (i < n) dst[i] = f32_to_bf16u(src[i]);
}

// f32 (rows x cols) -> bf16 into a wider row (dcols), at column offset 0
__global__ void cvt_f32_bf16_strided(const float* __restrict__ src, unsigned short* __restrict__ dst,
                                     int cols, int dcols, long n) {
    long i = (long)blockIdx.x * blockDim.x + threadIdx.x;
    if (i >= n) return;
    long r = i / cols; int c = (int)(i % cols);
    dst[r * (long)dcols + c] = f32_to_bf16u(src[i]);
}

// ---------------------------------------------------------------------------
// WMMA GEMM:  C[M,N] = A[M,K](bf16, lda) * W[N,K]^T (bf16, ldw)  (+resid)(+relu)
// Block: 128 threads = 4 waves (2Mx2N). Wave tile 32x64 (2 A-frags x 4 B-frags
// -> 8 WMMA per K-step, B fragments reused across both A fragments).
// Block tile 64x128. A/B fragment layouts per CDNA5 ISA 7.12.2.
// ---------------------------------------------------------------------------
template <int OUT_F32, int RELU, int RESID>
__global__ __launch_bounds__(128) void gemm_bf16_wmma(
    const unsigned short* __restrict__ A, int lda,
    const unsigned short* __restrict__ W, int ldw,
    void* __restrict__ Cout, int ldc,
    const float* __restrict__ resid,
    int M, int N, int K)
{
    const int lane  = threadIdx.x & 31;
    const int wave  = threadIdx.x >> 5;        // 0..3
    const int waveM = wave >> 1;               // 0..1
    const int waveN = wave & 1;                // 0..1
    const int bM = blockIdx.y * 64 + waveM * 32;
    const int bN = blockIdx.x * 128 + waveN * 64;
    const int half = lane >> 4;                // K half-select
    const int mr   = lane & 15;                // row within fragment

    const unsigned short* Ap[2];
#pragma unroll
    for (int mm = 0; mm < 2; ++mm) {
        int arow = bM + mm * 16 + mr;
        if (arow >= M) arow = M - 1;
        Ap[mm] = A + (size_t)arow * lda;
    }

    const unsigned short* Wp[4];
#pragma unroll
    for (int t = 0; t < 4; ++t) {
        int n = bN + t * 16 + mr;
        if (n >= N) n = N - 1;
        Wp[t] = W + (size_t)n * ldw;
    }

    v8f acc[2][4];
#pragma unroll
    for (int mm = 0; mm < 2; ++mm)
#pragma unroll
        for (int t = 0; t < 4; ++t)
#pragma unroll
            for (int r = 0; r < 8; ++r) acc[mm][t][r] = 0.f;

    for (int k = 0; k < K; k += 32) {
        __builtin_prefetch(Ap[0] + k + 128, 0, 1);  // global_prefetch_b8
        v16bf a[2];
#pragma unroll
        for (int mm = 0; mm < 2; ++mm) {
            const v8bf c0 = *reinterpret_cast<const v8bf*>(Ap[mm] + k + 8 * half);
            const v8bf c1 = *reinterpret_cast<const v8bf*>(Ap[mm] + k + 16 + 8 * half);
#pragma unroll
            for (int i = 0; i < 8; ++i) { a[mm][i] = c0[i]; a[mm][8 + i] = c1[i]; }
        }
#pragma unroll
        for (int t = 0; t < 4; ++t) {
            v16bf b;
            const v8bf c0 = *reinterpret_cast<const v8bf*>(Wp[t] + k + 8 * half);
            const v8bf c1 = *reinterpret_cast<const v8bf*>(Wp[t] + k + 16 + 8 * half);
#pragma unroll
            for (int i = 0; i < 8; ++i) { b[i] = c0[i]; b[8 + i] = c1[i]; }
#pragma unroll
            for (int mm = 0; mm < 2; ++mm) {
                acc[mm][t] = __builtin_amdgcn_wmma_f32_16x16x32_bf16(
                    false, a[mm], false, b, (short)0, acc[mm][t], false, false);
            }
        }
    }

    // C layout: VGPR r, lanes 0-15 -> M=r,N=lane ; lanes 16-31 -> M=8+r,N=lane-16
    const int cn0 = bN + (lane & 15);
#pragma unroll
    for (int mm = 0; mm < 2; ++mm) {
        const int cm = bM + mm * 16 + half * 8;
#pragma unroll
        for (int t = 0; t < 4; ++t) {
            int gn = cn0 + t * 16;
            if (gn >= N) continue;
#pragma unroll
            for (int r = 0; r < 8; ++r) {
                int gm = cm + r;
                if (gm >= M) continue;
                float v = acc[mm][t][r];
                if (RESID) v += resid[(size_t)gm * ldc + gn];
                if (RELU)  v = v > 0.f ? v : 0.f;
                if (OUT_F32) ((float*)Cout)[(size_t)gm * ldc + gn] = v;
                else ((unsigned short*)Cout)[(size_t)gm * ldc + gn] = f32_to_bf16u(v);
            }
        }
    }
}

// ---------------------------------------------------------------------------
// Input RMSNorm (1024-wide, eps 1e-4) -> bf16
// ---------------------------------------------------------------------------
__global__ __launch_bounds__(256) void rmsnorm_in_kernel(const float* __restrict__ x,
                                                         const float* __restrict__ w,
                                                         unsigned short* __restrict__ ubf)
{
    int row = blockIdx.x;
    const float* xr = x + (size_t)row * D_MODEL;
    float v[4]; float ss = 0.f;
#pragma unroll
    for (int i = 0; i < 4; ++i) { v[i] = xr[threadIdx.x + i * 256]; ss += v[i] * v[i]; }
    __shared__ float red[256];
    red[threadIdx.x] = ss; __syncthreads();
    for (int s = 128; s > 0; s >>= 1) { if (threadIdx.x < s) red[threadIdx.x] += red[threadIdx.x + s]; __syncthreads(); }
    float inv = rsqrtf(red[0] / (float)D_MODEL + 1e-4f);
#pragma unroll
    for (int i = 0; i < 4; ++i) {
        int c = threadIdx.x + i * 256;
        ubf[(size_t)row * D_MODEL + c] = f32_to_bf16u(v[i] * inv * w[c]);
    }
}

// ---------------------------------------------------------------------------
// dt: softplus(dt_raw + dt_bias), dA = exp(dt * -exp(A_log))
// ---------------------------------------------------------------------------
__global__ void dt_kernel(const unsigned short* __restrict__ zxbcdt,
                          const float* __restrict__ dt_bias, const float* __restrict__ A_log,
                          float* __restrict__ dtv, float* __restrict__ dAv)
{
    int idx = blockIdx.x * blockDim.x + threadIdx.x;   // BT*32
    if (idx >= BT * NHEADS) return;
    int h = idx & 31; int bt = idx >> 5;
    float v = bf16u_to_f32(zxbcdt[(size_t)bt * D_INPROJ + 2 * D_INNER + 2 * D_STATE + h]) + dt_bias[h];
    float sp = (v > 20.f) ? v : __logf(1.f + __expf(v));
    dtv[idx] = sp;
    dAv[idx] = __expf(-__expf(A_log[h]) * sp);
}

// ---------------------------------------------------------------------------
// Causal depthwise conv (window 4) + bias + silu -> bf16
// ---------------------------------------------------------------------------
__global__ void conv_silu_kernel(const unsigned short* __restrict__ zxbcdt,
                                 const float* __restrict__ conv_w, const float* __restrict__ conv_b,
                                 unsigned short* __restrict__ xBCc)
{
    long idx = (long)blockIdx.x * blockDim.x + threadIdx.x;   // B*T*2304
    if (idx >= (long)BB * TT * XBC_DIM) return;
    int c = (int)(idx % XBC_DIM);
    int t = (int)((idx / XBC_DIM) % TT);
    int b = (int)(idx / ((long)XBC_DIM * TT));
    const size_t base = ((size_t)b * TT) * D_INPROJ + D_INNER + c;
    float acc = conv_b[c];
#pragma unroll
    for (int j = 0; j < D_CONV; ++j) {
        int tt = t - (D_CONV - 1) + j;
        if (tt >= 0)
            acc += bf16u_to_f32(zxbcdt[base + (size_t)tt * D_INPROJ]) * conv_w[c * D_CONV + j];
    }
    xBCc[idx] = f32_to_bf16u(siluf_(acc));
}

// ---------------------------------------------------------------------------
// Selective scan. 1 block per (b,h). thread: p = tid>>2, owns 32 n-values.
// h state lives in VGPRs; y reduced across 4 lanes via shfl_xor.
// ---------------------------------------------------------------------------
__global__ __launch_bounds__(256) void scan_kernel(
    const unsigned short* __restrict__ xBCc,  // (B,T,2304) bf16
    const float* __restrict__ dtv,            // (B,T,32)
    const float* __restrict__ dAv,            // (B,T,32)
    const float* __restrict__ Dp,             // (32)
    float* __restrict__ yscan)                // (B,T,2048)
{
    const int blk = blockIdx.x;            // 0..127
    const int b = blk >> 5;
    const int h = blk & 31;
    const int tid = threadIdx.x;
    const int p  = tid >> 2;
    const int ng = tid & 3;
    const int nbase = ng * 32;
    float hreg[32];
#pragma unroll
    for (int i = 0; i < 32; ++i) hreg[i] = 0.f;
    const float dph = Dp[h];
    const size_t rowXBC = (size_t)b * TT * XBC_DIM;
    const size_t rowDT  = (size_t)b * TT * NHEADS + h;
    for (int t = 0; t < TT; ++t) {
        const unsigned short* xr = xBCc + rowXBC + (size_t)t * XBC_DIM;
        float xv = bf16u_to_f32(xr[h * HEADDIM + p]);
        float dA = dAv[rowDT + (size_t)t * NHEADS];
        float dt = dtv[rowDT + (size_t)t * NHEADS];
        float coef = dt * xv;
        const unsigned short* Br = xr + D_INNER + nbase;
        const unsigned short* Cr = xr + D_INNER + D_STATE + nbase;
        float accy = 0.f;
#pragma unroll
        for (int i = 0; i < 32; ++i) {
            float Bi = bf16u_to_f32(Br[i]);
            float Ci = bf16u_to_f32(Cr[i]);
            float hv = __fmaf_rn(dA, hreg[i], coef * Bi);
            hreg[i] = hv;
            accy = __fmaf_rn(hv, Ci, accy);
        }
        accy += __shfl_xor(accy, 1);
        accy += __shfl_xor(accy, 2);
        if (ng == 0)
            yscan[((size_t)b * TT + t) * D_INNER + h * HEADDIM + p] = accy + dph * xv;
    }
}

// ---------------------------------------------------------------------------
// ybar = rmsnorm(yscan * silu(z), ssm_norm_w, 1e-5)  -> bf16 (2048-wide)
// ---------------------------------------------------------------------------
__global__ __launch_bounds__(256) void ssm_norm_kernel(const float* __restrict__ yscan,
                                                       const unsigned short* __restrict__ zxbcdt,
                                                       const float* __restrict__ w,
                                                       unsigned short* __restrict__ ybar)
{
    int row = blockIdx.x;   // BT
    const float* yr = yscan + (size_t)row * D_INNER;
    const unsigned short* zr = zxbcdt + (size_t)row * D_INPROJ;
    float prod[8]; float ss = 0.f;
#pragma unroll
    for (int i = 0; i < 8; ++i) {
        int c = threadIdx.x + i * 256;
        float z = bf16u_to_f32(zr[c]);
        float pv = yr[c] * siluf_(z);
        prod[i] = pv; ss += pv * pv;
    }
    __shared__ float red[256];
    red[threadIdx.x] = ss; __syncthreads();
    for (int s = 128; s > 0; s >>= 1) { if (threadIdx.x < s) red[threadIdx.x] += red[threadIdx.x + s]; __syncthreads(); }
    float inv = rsqrtf(red[0] / (float)D_INNER + 1e-5f);
#pragma unroll
    for (int i = 0; i < 8; ++i) {
        int c = threadIdx.x + i * 256;
        ybar[(size_t)row * D_INNER + c] = f32_to_bf16u(prod[i] * inv * w[c]);
    }
}

// ---------------------------------------------------------------------------
// scores = sigmoid(dot(relu_hidden, score_w2))
// ---------------------------------------------------------------------------
__global__ __launch_bounds__(256) void score2_kernel(const float* __restrict__ s1,
                                                     const float* __restrict__ w2,
                                                     float* __restrict__ scores)
{
    int row = blockIdx.x;
    __shared__ float red[256];
    red[threadIdx.x] = s1[(size_t)row * SUMDIM + threadIdx.x] * w2[threadIdx.x];
    __syncthreads();
    for (int s = 128; s > 0; s >>= 1) { if (threadIdx.x < s) red[threadIdx.x] += red[threadIdx.x + s]; __syncthreads(); }
    if (threadIdx.x == 0) scores[row] = sigmoidf_(red[0]);
}

// ---------------------------------------------------------------------------
// Top-64 selection (stable argsort-descending semantics, ties -> lowest idx)
// ---------------------------------------------------------------------------
__global__ __launch_bounds__(256) void topk_kernel(const float* __restrict__ scores,
                                                   int* __restrict__ order, float* __restrict__ topscore)
{
    int b = blockIdx.x;
    __shared__ float sc[TT];
    for (int i = threadIdx.x; i < TT; i += 256) sc[i] = scores[b * TT + i];
    __syncthreads();
    __shared__ float bv_[256]; __shared__ int bi_[256];
    for (int sel = 0; sel < POOLN; ++sel) {
        float bv = -1e30f; int bi = 0;
        for (int i = threadIdx.x; i < TT; i += 256) {
            float v = sc[i];
            if (v > bv) { bv = v; bi = i; }
        }
        bv_[threadIdx.x] = bv; bi_[threadIdx.x] = bi; __syncthreads();
        for (int s = 128; s > 0; s >>= 1) {
            if (threadIdx.x < s) {
                if (bv_[threadIdx.x + s] > bv_[threadIdx.x] ||
                    (bv_[threadIdx.x + s] == bv_[threadIdx.x] && bi_[threadIdx.x + s] < bi_[threadIdx.x])) {
                    bv_[threadIdx.x] = bv_[threadIdx.x + s]; bi_[threadIdx.x] = bi_[threadIdx.x + s];
                }
            }
            __syncthreads();
        }
        if (threadIdx.x == 0) {
            order[b * POOLN + sel] = bi_[0];
            topscore[b * POOLN + sel] = bv_[0];
            sc[bi_[0]] = -1e30f;
        }
        __syncthreads();
    }
}

// ---------------------------------------------------------------------------
// summaries[b,i,:] = y[b, order[b,i], :] @ summ_w^T
// ---------------------------------------------------------------------------
__global__ __launch_bounds__(256) void summary_kernel(const float* __restrict__ y,
                                                      const int* __restrict__ order,
                                                      const float* __restrict__ summ_w,
                                                      float* __restrict__ summ)
{
    int b = blockIdx.x >> 6; int i = blockIdx.x & 63;
    int tok = order[b * POOLN + i];
    __shared__ float row[D_MODEL];
    const float* yr = y + ((size_t)b * TT + tok) * D_MODEL;
    for (int c = threadIdx.x; c < D_MODEL; c += 256) row[c] = yr[c];
    __syncthreads();
    const float* wr = summ_w + (size_t)threadIdx.x * D_MODEL;
    float acc = 0.f;
    for (int c = 0; c < D_MODEL; ++c) acc = __fmaf_rn(row[c], wr[c], acc);
    summ[((size_t)b * POOLN + i) * SUMDIM + threadIdx.x] = acc;
}

// ---------------------------------------------------------------------------
// Sequential 64-slot pool update (reference add-then-replace semantics)
// ---------------------------------------------------------------------------
__global__ __launch_bounds__(64) void pool_scan_kernel(const float* __restrict__ topscore,
                                                       const float* __restrict__ summ,
                                                       float* __restrict__ pool_o,
                                                       float* __restrict__ pri_o,
                                                       int* __restrict__ counts_io)
{
    int b = blockIdx.x;
    int tid = threadIdx.x;
    __shared__ float pri_s[64];
    __shared__ int cnt_s;
    __shared__ float redv[64]; __shared__ int redi[64];
    __shared__ int s_widx, s_do_add, s_do_rep, s_minidx;
    pri_s[tid] = pri_o[b * POOLN + tid];
    if (tid == 0) cnt_s = counts_io[b];
    __syncthreads();
    for (int sl = 0; sl < POOLN; ++sl) {
        float imp = topscore[b * POOLN + sl];
        int has = imp > 0.5f;   // TAU1
        if (tid == 0) {
            int cnt = cnt_s;
            int add = has && (cnt < POOLN);
            int widx = cnt < POOLN - 1 ? cnt : POOLN - 1;
            s_widx = widx; s_do_add = add;
            if (add) { pri_s[widx] = imp; cnt_s = cnt + 1; }
            s_do_rep = has && (cnt_s >= POOLN);
        }
        __syncthreads();
        if (s_do_add) {
            const float* sr = summ + ((size_t)b * POOLN + sl) * SUMDIM;
            float* pr = pool_o + ((size_t)b * POOLN + s_widx) * SUMDIM;
            for (int c = tid; c < SUMDIM; c += 64) pr[c] = sr[c];
        }
        __syncthreads();
        if (s_do_rep) {
            redv[tid] = pri_s[tid]; redi[tid] = tid; __syncthreads();
            for (int s = 32; s > 0; s >>= 1) {
                if (tid < s) {
                    if (redv[tid + s] < redv[tid] ||
                        (redv[tid + s] == redv[tid] && redi[tid + s] < redi[tid])) {
                        redv[tid] = redv[tid + s]; redi[tid] = redi[tid + s];
                    }
                }
                __syncthreads();
            }
            if (tid == 0) {
                s_minidx = redi[0];
                s_do_rep = (imp > redv[0]);
                if (s_do_rep) pri_s[s_minidx] = imp;
            }
            __syncthreads();
            if (s_do_rep) {
                const float* sr = summ + ((size_t)b * POOLN + sl) * SUMDIM;
                float* pr = pool_o + ((size_t)b * POOLN + s_minidx) * SUMDIM;
                for (int c = tid; c < SUMDIM; c += 64) pr[c] = sr[c];
            }
        }
        __syncthreads();
    }
    pri_o[b * POOLN + tid] = pri_s[tid];
    if (tid == 0) counts_io[b] = cnt_s;
}

// ---------------------------------------------------------------------------
// rmask per batch + float counts output
// ---------------------------------------------------------------------------
__global__ __launch_bounds__(256) void rmask_kernel(const float* __restrict__ scores,
                                                    const int* __restrict__ cnt,
                                                    float* __restrict__ rmaskf,
                                                    float* __restrict__ out_counts)
{
    int b = blockIdx.x;
    __shared__ float red[256];
    float s = 0.f;
    for (int i = threadIdx.x; i < TT; i += 256) s += scores[b * TT + i];
    red[threadIdx.x] = s; __syncthreads();
    for (int st = 128; st > 0; st >>= 1) { if (threadIdx.x < st) red[threadIdx.x] += red[threadIdx.x + st]; __syncthreads(); }
    if (threadIdx.x == 0) {
        float mean = red[0] / (float)TT;
        rmaskf[b] = (mean > 0.4f && cnt[b] > 0) ? 1.f : 0.f;   // TAU2
        out_counts[b] = (float)cnt[b];
    }
}

// ---------------------------------------------------------------------------
// k = pool @ k_w^T , v = pool @ v_w^T   (per pool row)
// ---------------------------------------------------------------------------
__global__ __launch_bounds__(256) void kv_kernel(const float* __restrict__ pool_o,
                                                 const float* __restrict__ k_w,
                                                 const float* __restrict__ v_w,
                                                 float* __restrict__ kbuf,
                                                 float* __restrict__ vbuf)
{
    int bp = blockIdx.x;   // b*64+p
    __shared__ float row[SUMDIM];
    row[threadIdx.x] = pool_o[(size_t)bp * SUMDIM + threadIdx.x];
    __syncthreads();
    {
        const float* wr = k_w + (size_t)threadIdx.x * SUMDIM;
        float acc = 0.f;
        for (int c = 0; c < SUMDIM; ++c) acc = __fmaf_rn(row[c], wr[c], acc);
        kbuf[(size_t)bp * SUMDIM + threadIdx.x] = acc;
    }
#pragma unroll
    for (int i = 0; i < 4; ++i) {
        int d = threadIdx.x + i * 256;
        const float* wr = v_w + (size_t)d * SUMDIM;
        float acc = 0.f;
        for (int c = 0; c < SUMDIM; ++c) acc = __fmaf_rn(row[c], wr[c], acc);
        vbuf[(size_t)bp * D_MODEL + d] = acc;
    }
}

// ---------------------------------------------------------------------------
// Pool attention: per (b,t), 64 slots, masked softmax, retrieved -> f32 + bf16
// ---------------------------------------------------------------------------
__global__ __launch_bounds__(256) void attn_kernel(const float* __restrict__ q,
                                                   const float* __restrict__ kbuf,
                                                   const float* __restrict__ vbuf,
                                                   const int* __restrict__ counts,
                                                   float* __restrict__ retr,
                                                   unsigned short* __restrict__ catbf)
{
    int bt = blockIdx.x; int b = bt >> 11;
    __shared__ float qrow[SUMDIM]; __shared__ float wts[POOLN];
    qrow[threadIdx.x] = q[(size_t)bt * SUMDIM + threadIdx.x];
    __syncthreads();
    int cnt = counts[b];
    if (threadIdx.x < POOLN) {
        int p = threadIdx.x;
        const float* kr = kbuf + ((size_t)b * POOLN + p) * SUMDIM;
        float acc = 0.f;
        for (int c = 0; c < SUMDIM; ++c) acc = __fmaf_rn(qrow[c], kr[c], acc);
        wts[p] = acc * 0.0625f;   // 1/sqrt(256)
    }
    __syncthreads();
    if (threadIdx.x == 0) {
        float mx = -1e30f;
        for (int p = 0; p < cnt && p < POOLN; ++p) mx = fmaxf(mx, wts[p]);
        float sum = 0.f;
        for (int p = 0; p < POOLN; ++p) {
            float w = (p < cnt) ? __expf(wts[p] - mx) : 0.f;
            wts[p] = w; sum += w;
        }
        float inv = (sum > 0.f) ? 1.f / sum : 0.f;
        for (int p = 0; p < POOLN; ++p) wts[p] *= inv;
    }
    __syncthreads();
#pragma unroll
    for (int i = 0; i < 4; ++i) {
        int d = threadIdx.x + i * 256;
        float acc = 0.f;
        for (int p = 0; p < POOLN; ++p)
            acc = __fmaf_rn(wts[p], vbuf[((size_t)b * POOLN + p) * D_MODEL + d], acc);
        retr[(size_t)bt * D_MODEL + d] = acc;
        catbf[(size_t)bt * (2 * D_MODEL) + D_MODEL + d] = f32_to_bf16u(acc);
    }
}

// ---------------------------------------------------------------------------
// Final: out = y + sigmoid(gate) * retrieved * rmask[b]
// ---------------------------------------------------------------------------
__global__ void final_kernel(const float* __restrict__ y, const float* __restrict__ gate,
                             const float* __restrict__ retr, const float* __restrict__ rmaskf,
                             float* __restrict__ out_y)
{
    long idx = (long)blockIdx.x * blockDim.x + threadIdx.x;   // BT*1024
    if (idx >= (long)BT * D_MODEL) return;
    int b = (int)(idx >> 21);
    out_y[idx] = y[idx] + sigmoidf_(gate[idx]) * retr[idx] * rmaskf[b];
}

// ---------------------------------------------------------------------------
// Host launch
// ---------------------------------------------------------------------------
static inline size_t align256(size_t x) { return (x + 255) & ~((size_t)255); }

extern "C" void kernel_launch(void* const* d_in, const int* in_sizes, int n_in,
                              void* d_out, int out_size, void* d_ws, size_t ws_size,
                              hipStream_t stream)
{
    (void)in_sizes; (void)n_in; (void)out_size; (void)ws_size;

    const float* x        = (const float*)d_in[0];
    const float* pool_in  = (const float*)d_in[1];
    const float* pri_in   = (const float*)d_in[2];
    const int*   counts_in= (const int*)  d_in[3];
    const float* norm_w   = (const float*)d_in[4];
    const float* in_w     = (const float*)d_in[5];
    const float* conv_w   = (const float*)d_in[6];
    const float* conv_b   = (const float*)d_in[7];
    const float* dt_bias  = (const float*)d_in[8];
    const float* A_log    = (const float*)d_in[9];
    const float* Dp       = (const float*)d_in[10];
    const float* ssm_norm_w=(const float*)d_in[11];
    const float* out_w    = (const float*)d_in[12];
    const float* score_w1 = (const float*)d_in[13];
    const float* score_w2 = (const float*)d_in[14];
    const float* summ_w   = (const float*)d_in[15];
    const float* q_w      = (const float*)d_in[16];
    const float* k_w      = (const float*)d_in[17];
    const float* v_w      = (const float*)d_in[18];
    const float* gate_w   = (const float*)d_in[19];

    // output layout: y | pool | priorities | counts(float)
    float* outY    = (float*)d_out;
    float* outPool = outY + (size_t)BT * D_MODEL;
    float* outPri  = outPool + (size_t)BB * POOLN * SUMDIM;
    float* outCnt  = outPri + (size_t)BB * POOLN;

    // workspace carve-up
    char* ws = (char*)d_ws;
    size_t off = 0;
    auto alloc = [&](size_t bytes) -> void* { void* p = ws + off; off = align256(off + bytes); return p; };

    unsigned short* in_w_bf   = (unsigned short*)alloc((size_t)D_INPROJ * D_MODEL * 2);
    unsigned short* out_w_bf  = (unsigned short*)alloc((size_t)D_MODEL * D_INNER * 2);
    unsigned short* sw1_bf    = (unsigned short*)alloc((size_t)SUMDIM * D_MODEL * 2);
    unsigned short* q_w_bf    = (unsigned short*)alloc((size_t)SUMDIM * D_MODEL * 2);
    unsigned short* gate_w_bf = (unsigned short*)alloc((size_t)D_MODEL * (2 * D_MODEL) * 2);
    unsigned short* ubf       = (unsigned short*)alloc((size_t)BT * D_MODEL * 2);
    unsigned short* zxbcdt    = (unsigned short*)alloc((size_t)BT * D_INPROJ * 2);
    unsigned short* xBCc      = (unsigned short*)alloc((size_t)BT * XBC_DIM * 2);
    float* dtv    = (float*)alloc((size_t)BT * NHEADS * 4);
    float* dAv    = (float*)alloc((size_t)BT * NHEADS * 4);
    float* yscan  = (float*)alloc((size_t)BT * D_INNER * 4);
    unsigned short* ybar = (unsigned short*)alloc((size_t)BT * D_INNER * 2);
    float* ybuf   = (float*)alloc((size_t)BT * D_MODEL * 4);
    unsigned short* catbf = (unsigned short*)alloc((size_t)BT * 2 * D_MODEL * 2);
    float* s1     = (float*)alloc((size_t)BT * SUMDIM * 4);
    float* scores = (float*)alloc((size_t)BT * 4);
    float* qbuf   = (float*)alloc((size_t)BT * SUMDIM * 4);
    int*   order  = (int*)alloc((size_t)BB * POOLN * 4);
    float* topsc  = (float*)alloc((size_t)BB * POOLN * 4);
    float* summ   = (float*)alloc((size_t)BB * POOLN * SUMDIM * 4);
    int*   cnts   = (int*)alloc(BB * 4);
    float* rmaskf = (float*)alloc(BB * 4);
    float* kbuf   = (float*)alloc((size_t)BB * POOLN * SUMDIM * 4);
    float* vbuf   = (float*)alloc((size_t)BB * POOLN * D_MODEL * 4);
    float* retr   = (float*)alloc((size_t)BT * D_MODEL * 4);
    float* gate   = (float*)alloc((size_t)BT * D_MODEL * 4);

    auto cvt = [&](const float* s, unsigned short* d, long n) {
        cvt_f32_bf16<<<(unsigned)((n + 255) / 256), 256, 0, stream>>>(s, d, n);
    };
    // weight conversions
    cvt(in_w,    in_w_bf,   (long)D_INPROJ * D_MODEL);
    cvt(out_w,   out_w_bf,  (long)D_MODEL * D_INNER);
    cvt(score_w1,sw1_bf,    (long)SUMDIM * D_MODEL);
    cvt(q_w,     q_w_bf,    (long)SUMDIM * D_MODEL);
    cvt(gate_w,  gate_w_bf, (long)D_MODEL * 2 * D_MODEL);

    // 1. input rmsnorm -> bf16
    rmsnorm_in_kernel<<<BT, 256, 0, stream>>>(x, norm_w, ubf);

    // 2. in_proj GEMM: (8192 x 4384 x 1024), bf16 out
    gemm_bf16_wmma<0, 0, 0><<<dim3((D_INPROJ + 127) / 128, BT / 64), 128, 0, stream>>>(
        ubf, D_MODEL, in_w_bf, D_MODEL, zxbcdt, D_INPROJ, nullptr, BT, D_INPROJ, D_MODEL);

    // 3. dt / dA
    dt_kernel<<<(BT * NHEADS + 255) / 256, 256, 0, stream>>>(zxbcdt, dt_bias, A_log, dtv, dAv);

    // 4. causal conv + silu
    {
        long n = (long)BT * XBC_DIM;
        conv_silu_kernel<<<(unsigned)((n + 255) / 256), 256, 0, stream>>>(zxbcdt, conv_w, conv_b, xBCc);
    }

    // 5. selective scan
    scan_kernel<<<BB * NHEADS, 256, 0, stream>>>(xBCc, dtv, dAv, Dp, yscan);

    // 6. gated rmsnorm -> ybar bf16
    ssm_norm_kernel<<<BT, 256, 0, stream>>>(yscan, zxbcdt, ssm_norm_w, ybar);

    // 7. out_proj GEMM + residual x -> y f32
    gemm_bf16_wmma<1, 0, 1><<<dim3(D_MODEL / 128, BT / 64), 128, 0, stream>>>(
        ybar, D_INNER, out_w_bf, D_INNER, ybuf, D_MODEL, x, BT, D_MODEL, D_INNER);

    // 8. y -> bf16 into first half of concat buffer
    {
        long n = (long)BT * D_MODEL;
        cvt_f32_bf16_strided<<<(unsigned)((n + 255) / 256), 256, 0, stream>>>(
            ybuf, catbf, D_MODEL, 2 * D_MODEL, n);
    }

    // 9. score hidden GEMM (relu epilogue)
    gemm_bf16_wmma<1, 1, 0><<<dim3(SUMDIM / 128, BT / 64), 128, 0, stream>>>(
        catbf, 2 * D_MODEL, sw1_bf, D_MODEL, s1, SUMDIM, nullptr, BT, SUMDIM, D_MODEL);

    // 10. scores
    score2_kernel<<<BT, 256, 0, stream>>>(s1, score_w2, scores);

    // 11. q GEMM
    gemm_bf16_wmma<1, 0, 0><<<dim3(SUMDIM / 128, BT / 64), 128, 0, stream>>>(
        catbf, 2 * D_MODEL, q_w_bf, D_MODEL, qbuf, SUMDIM, nullptr, BT, SUMDIM, D_MODEL);

    // 12. pool state: copy inputs into output regions / workspace
    (void)hipMemcpyAsync(outPool, pool_in, (size_t)BB * POOLN * SUMDIM * 4, hipMemcpyDeviceToDevice, stream);
    (void)hipMemcpyAsync(outPri,  pri_in,  (size_t)BB * POOLN * 4,          hipMemcpyDeviceToDevice, stream);
    (void)hipMemcpyAsync(cnts,    counts_in, BB * 4,                        hipMemcpyDeviceToDevice, stream);

    // 13. top-64, summaries, sequential pool update
    topk_kernel<<<BB, 256, 0, stream>>>(scores, order, topsc);
    summary_kernel<<<BB * POOLN, 256, 0, stream>>>(ybuf, order, summ_w, summ);
    pool_scan_kernel<<<BB, 64, 0, stream>>>(topsc, summ, outPool, outPri, cnts);

    // 14. rmask + float counts
    rmask_kernel<<<BB, 256, 0, stream>>>(scores, cnts, rmaskf, outCnt);

    // 15. k/v from updated pool
    kv_kernel<<<BB * POOLN, 256, 0, stream>>>(outPool, k_w, v_w, kbuf, vbuf);

    // 16. pool attention -> retrieved (f32 + bf16 into concat)
    attn_kernel<<<BT, 256, 0, stream>>>(qbuf, kbuf, vbuf, cnts, retr, catbf);

    // 17. gate GEMM over [y, retrieved]
    gemm_bf16_wmma<1, 0, 0><<<dim3(D_MODEL / 128, BT / 64), 128, 0, stream>>>(
        catbf, 2 * D_MODEL, gate_w_bf, 2 * D_MODEL, gate, D_MODEL, nullptr, BT, D_MODEL, 2 * D_MODEL);

    // 18. final gated residual -> output y
    {
        long n = (long)BT * D_MODEL;
        final_kernel<<<(unsigned)((n + 255) / 256), 256, 0, stream>>>(ybuf, gate, retr, rmaskf, outY);
    }
}
